// DualCrossAttention_49005576847529
// MI455X (gfx1250) — matrix operand order repaired and checked by reading, not compiled, not run on hardware
//
#include <hip/hip_runtime.h>
#include <math.h>

typedef __bf16 bf16;
typedef __attribute__((ext_vector_type(16))) __bf16 v16bf;
typedef __attribute__((ext_vector_type(8)))  __bf16 v8bf;
typedef __attribute__((ext_vector_type(8)))  float  v8f;
typedef __attribute__((ext_vector_type(4))) unsigned int u32x4;
typedef __attribute__((ext_vector_type(8))) int i32x8;
typedef __attribute__((ext_vector_type(4))) int i32x4;

static __device__ inline v16bf pack16(v8bf lo, v8bf hi) {
  v16bf r;
#pragma unroll
  for (int i = 0; i < 8; ++i) { r[i] = lo[i]; r[8 + i] = hi[i]; }
  return r;
}

// ---------------------------------------------------------------------------
// Tensor Data Mover: 2D tile (tile_d0 x tile_d1, bf16) from global to LDS,
// with 16B of LDS padding appended after every row (bank-conflict-free
// strides for 16-lane x b128 fragment reads).
// pad_code: row length in DWORDs = 2^(pad_code+1)  (16 DW -> 3, 32 DW -> 4)
// Issued once per wave (EXEC ignored by TDM); tracked by TENSORcnt.
// ---------------------------------------------------------------------------
static __device__ inline void tdm_load_2d(unsigned lds_addr, const void* gaddr,
                                          unsigned tensor_d0, unsigned tensor_d1,
                                          unsigned long long stride0,
                                          unsigned tile_d0, unsigned tile_d1,
                                          unsigned pad_code) {
  unsigned long long ga = (unsigned long long)gaddr;
  u32x4 g0;
  g0[0] = 1u;                                    // count=1 valid descriptor
  g0[1] = lds_addr;                              // LDS byte offset
  g0[2] = (unsigned)(ga & 0xffffffffu);          // global addr [31:0]
  g0[3] = (unsigned)((ga >> 32) & 0x1ffffffu)    // global addr [56:32]
          | 0x80000000u;                         // type = 2 ("image")
  i32x8 g1;
  unsigned w0 = (1u << 16)                       // data_size = 2 bytes (bf16)
              | (1u << 20)                       // pad_enable
              | (pad_code << 22)                 // pad_interval
              | (3u << 25);                      // pad_amount = 4 DWORDs (16B)
  g1[0] = (int)w0;
  g1[1] = (int)((tensor_d0 & 0xffffu) << 16);            // tensor_dim0[15:0]
  g1[2] = (int)((tensor_d0 >> 16) | ((tensor_d1 & 0xffffu) << 16));
  g1[3] = (int)((tensor_d1 >> 16) | (tile_d0 << 16));    // tile_dim0
  g1[4] = (int)tile_d1;                                  // tile_dim1, tile_dim2=0
  g1[5] = (int)(unsigned)(stride0 & 0xffffffffull);      // dim0 stride lo
  g1[6] = (int)(unsigned)((stride0 >> 32) & 0xffffull);  // dim0 stride hi
  g1[7] = 0;
  i32x4 z4 = {0, 0, 0, 0};
#if defined(__clang_major__) && __clang_major__ >= 23
  i32x8 z8 = {0, 0, 0, 0, 0, 0, 0, 0};
  __builtin_amdgcn_tensor_load_to_lds(g0, g1, z4, z4, z8, 0);
#else
  __builtin_amdgcn_tensor_load_to_lds(g0, g1, z4, z4, 0);
#endif
}

static __device__ inline unsigned lds_off(const void* p) {
  // generic pointer to LDS = {SHARED_BASE, offset[31:0]} -> low 32 bits
  return (unsigned)(unsigned long long)(uintptr_t)p;
}

// ---------------------------------------------------------------------------
// f32 -> bf16 elementwise convert
// ---------------------------------------------------------------------------
__global__ void cvt_f32_bf16(const float* __restrict__ x, bf16* __restrict__ y, size_t n) {
  size_t i = (size_t)blockIdx.x * blockDim.x + threadIdx.x;
  size_t stride = (size_t)gridDim.x * blockDim.x;
  for (; i < n; i += stride) y[i] = (bf16)x[i];
}

// ---------------------------------------------------------------------------
// V (B*S2, H*64) bf16 -> VT (B, H, 64, S2) bf16  (s contiguous)
// ---------------------------------------------------------------------------
__global__ void transpose_v(const bf16* __restrict__ v, bf16* __restrict__ vt, int S2) {
  size_t total = (size_t)8 * 8 * 64 * S2;
  size_t i = (size_t)blockIdx.x * blockDim.x + threadIdx.x;
  size_t stride = (size_t)gridDim.x * blockDim.x;
  for (; i < total; i += stride) {
    int s = (int)(i % S2);
    int d = (int)((i / S2) % 64);
    int h = (int)((i / ((size_t)S2 * 64)) % 8);
    int b = (int)(i / ((size_t)S2 * 64 * 8));
    vt[i] = v[((size_t)(b * S2 + s)) * 512 + h * 64 + d];
  }
}

// ---------------------------------------------------------------------------
// GEMM: out[M,N] = A[M,K] @ W[N,K]^T (+bias), bf16, f32 accumulate.
// Block tile 256x64 (8 waves x 32 rows, 4 N-tiles), K-step 32.
// TDM double-buffers the 256x32 A tile and 64x32 W tile in LDS (padded rows,
// 40 bf16 stride); all waves read fragments with ds_load_b128.
// grid = (N/64, M/256), block = 256. Requires M%256==0, N%64==0, K%32==0.
// ---------------------------------------------------------------------------
__global__ __launch_bounds__(256)
void gemm_bf16_wmma(const bf16* __restrict__ A, const bf16* __restrict__ W,
                    const float* __restrict__ bias,
                    float* __restrict__ outF, bf16* __restrict__ outB,
                    int M, int N, int K) {
  __shared__ bf16 ldsA[2][256 * 40];   // 32 elems + 8 pad per row
  __shared__ bf16 ldsW[2][64 * 40];
  const int lane = threadIdx.x & 31;
  const int wave = threadIdx.x >> 5;
  const int half = lane >> 4;
  const int l16  = lane & 15;
  const int m0 = blockIdx.y * 256;
  const int n0 = blockIdx.x * 64;
  const int mw = m0 + wave * 32;

  v8f zero8 = {0.f, 0.f, 0.f, 0.f, 0.f, 0.f, 0.f, 0.f};
  v8f acc[2][4];
#pragma unroll
  for (int mt = 0; mt < 2; ++mt)
#pragma unroll
    for (int nt = 0; nt < 4; ++nt) acc[mt][nt] = zero8;

  const int nsteps = K >> 5;
  if (wave == 0) {
    tdm_load_2d(lds_off(&ldsA[0][0]), A + (size_t)m0 * K, K, M, (unsigned long long)K, 32, 256, 3);
    tdm_load_2d(lds_off(&ldsW[0][0]), W + (size_t)n0 * K, K, N, (unsigned long long)K, 32, 64, 3);
  }
  int cur = 0;
  for (int step = 0; step < nsteps; ++step) {
    const bool has_next = (step + 1) < nsteps;
    if (wave == 0 && has_next) {
      int k0 = (step + 1) * 32;
      tdm_load_2d(lds_off(&ldsA[cur ^ 1][0]), A + (size_t)m0 * K + k0, K, M,
                  (unsigned long long)K, 32, 256, 3);
      tdm_load_2d(lds_off(&ldsW[cur ^ 1][0]), W + (size_t)n0 * K + k0, K, N,
                  (unsigned long long)K, 32, 64, 3);
    }
    if (has_next) __builtin_amdgcn_s_wait_tensorcnt(2);   // current buffer done
    else          __builtin_amdgcn_s_wait_tensorcnt(0);
    __syncthreads();

    v16bf a[2];
#pragma unroll
    for (int mt = 0; mt < 2; ++mt) {
      const bf16* ap = &ldsA[cur][(wave * 32 + mt * 16 + l16) * 40 + 8 * half];
      v8bf lo = *(const v8bf*)ap;
      v8bf hi = *(const v8bf*)(ap + 16);
      a[mt] = pack16(lo, hi);
    }
#pragma unroll
    for (int nt = 0; nt < 4; ++nt) {
      const bf16* wp = &ldsW[cur][(nt * 16 + l16) * 40 + 8 * half];
      v8bf blo = *(const v8bf*)wp;
      v8bf bhi = *(const v8bf*)(wp + 16);
      v16bf b = pack16(blo, bhi);
      acc[0][nt] = __builtin_amdgcn_wmma_f32_16x16x32_bf16(
          false, a[0], false, b, (short)0, acc[0][nt], false, false);
      acc[1][nt] = __builtin_amdgcn_wmma_f32_16x16x32_bf16(
          false, a[1], false, b, (short)0, acc[1][nt], false, false);
    }
    __syncthreads();   // everyone done reading before the buffer is refilled
    cur ^= 1;
  }

#pragma unroll
  for (int mt = 0; mt < 2; ++mt)
#pragma unroll
    for (int nt = 0; nt < 4; ++nt) {
      int n = n0 + nt * 16 + l16;
      float bv = bias ? bias[n] : 0.f;
#pragma unroll
      for (int r = 0; r < 8; ++r) {
        int m = mw + mt * 16 + r + 8 * half;
        float v = acc[mt][nt][r] + bv;
        size_t idx = (size_t)m * N + n;
        if (outF) outF[idx] = v;
        if (outB) outB[idx] = (bf16)v;
      }
    }
}

// ---------------------------------------------------------------------------
// Flash-style masked cross attention. 4 waves/block, each wave owns one
// 16-query tile; the 32-key K tile (32x64) and V^T tile (64x32) are shared
// through TDM double-buffered LDS. grid = (S1/64, H, B), block = 128.
// ---------------------------------------------------------------------------
__device__ inline float maskw(int mode, int q, int s) {
  if (mode == 1) return ((q < 512) == (s < 512)) ? 1.25f : 1.0f;
  return ((q >> 8) == (s >> 8)) ? 1.25f : 1.0f;
}

__global__ __launch_bounds__(128)
void attn_wmma(const bf16* __restrict__ Q, const bf16* __restrict__ Kb,
               const bf16* __restrict__ VT, bf16* __restrict__ ctx,
               int S1, int S2, int mode) {
  __shared__ bf16 ldsK[2][32 * 72];   // 64 elems + 8 pad per key row
  __shared__ bf16 ldsV[2][64 * 40];   // 32 elems + 8 pad per d row
  __shared__ bf16 pbuf[4][16 * 32];
  const int lane = threadIdx.x & 31;
  const int wave = threadIdx.x >> 5;
  const int half = lane >> 4;
  const int l16  = lane & 15;
  const int q0 = (blockIdx.x * 4 + wave) * 16;
  const int h  = blockIdx.y;
  const int b  = blockIdx.z;
  const float scale = 0.125f; // 1/sqrt(64)
  bf16* pb = pbuf[wave];

  // Q A-fragments for the two 32-wide K-dim steps (d = 0..31, 32..63)
  v16bf qf[2];
#pragma unroll
  for (int kk = 0; kk < 2; ++kk) {
    const bf16* qp = Q + (size_t)(b * S1 + q0 + l16) * 512 + h * 64 + kk * 32 + 8 * half;
    v8bf lo = *(const v8bf*)qp;
    v8bf hi = *(const v8bf*)(qp + 16);
    qf[kk] = pack16(lo, hi);
  }

  v8f zero8 = {0.f, 0.f, 0.f, 0.f, 0.f, 0.f, 0.f, 0.f};
  v8f o[4];
#pragma unroll
  for (int nt = 0; nt < 4; ++nt) o[nt] = zero8;
  float mrow[8], lrow[8];
#pragma unroll
  for (int r = 0; r < 8; ++r) { mrow[r] = -1e30f; lrow[r] = 0.f; }

  const bf16* kbase  = Kb + (size_t)b * S2 * 512 + h * 64;        // row stride 512
  const bf16* vtbase = VT + (size_t)((b * 8 + h) * 64) * S2;      // row stride S2
  const int nsteps = S2 >> 5;

  if (wave == 0) {
    tdm_load_2d(lds_off(&ldsK[0][0]), kbase, 64, S2, 512ull, 64, 32, 4);
    tdm_load_2d(lds_off(&ldsV[0][0]), vtbase, S2, 64, (unsigned long long)S2, 32, 64, 3);
  }
  int cur = 0;
  for (int step = 0; step < nsteps; ++step) {
    const int s0 = step * 32;
    const bool has_next = (step + 1) < nsteps;
    if (wave == 0 && has_next) {
      int s1 = s0 + 32;
      tdm_load_2d(lds_off(&ldsK[cur ^ 1][0]), kbase + (size_t)s1 * 512, 64, S2,
                  512ull, 64, 32, 4);
      tdm_load_2d(lds_off(&ldsV[cur ^ 1][0]), vtbase + s1, S2, 64,
                  (unsigned long long)S2, 32, 64, 3);
    }
    if (has_next) __builtin_amdgcn_s_wait_tensorcnt(2);
    else          __builtin_amdgcn_s_wait_tensorcnt(0);
    __syncthreads();

    // scores: two 16x16 subtiles over 32 keys, head-dim 64 = 2 WMMA k-steps
    v8f sc[2];
    sc[0] = zero8; sc[1] = zero8;
#pragma unroll
    for (int ss = 0; ss < 2; ++ss) {
#pragma unroll
      for (int kk = 0; kk < 2; ++kk) {
        const bf16* kp = &ldsK[cur][(ss * 16 + l16) * 72 + kk * 32 + 8 * half];
        v8bf klo = *(const v8bf*)kp;
        v8bf khi = *(const v8bf*)(kp + 16);
        v16bf kf = pack16(klo, khi);
        sc[ss] = __builtin_amdgcn_wmma_f32_16x16x32_bf16(
            false, qf[kk], false, kf, (short)0, sc[ss], false, false);
      }
    }
    // online softmax update per row (row = r + 8*half, stats replicated per half)
#pragma unroll
    for (int r = 0; r < 8; ++r) {
      int qg = q0 + r + 8 * half;
      int sg0 = s0 + l16, sg1 = s0 + 16 + l16;
      float v0 = sc[0][r] * (scale * maskw(mode, qg, sg0));
      float v1 = sc[1][r] * (scale * maskw(mode, qg, sg1));
      float mx = fmaxf(v0, v1);
#pragma unroll
      for (int d = 1; d < 16; d <<= 1) mx = fmaxf(mx, __shfl_xor(mx, d, 32));
      float mnew = fmaxf(mrow[r], mx);
      float p0 = __expf(v0 - mnew);
      float p1 = __expf(v1 - mnew);
      float rs = p0 + p1;
#pragma unroll
      for (int d = 1; d < 16; d <<= 1) rs += __shfl_xor(rs, d, 32);
      float f = __expf(mrow[r] - mnew);
      lrow[r] = lrow[r] * f + rs;
      mrow[r] = mnew;
#pragma unroll
      for (int nt = 0; nt < 4; ++nt) o[nt][r] *= f;
      pb[(r + 8 * half) * 32 + l16]      = (bf16)p0;
      pb[(r + 8 * half) * 32 + 16 + l16] = (bf16)p1;
    }
    // repack P (16x32) from C-layout via LDS into A-fragment layout
    v16bf pf;
    {
      const bf16* lp = &pb[l16 * 32 + 8 * half];
      v8bf lo = *(const v8bf*)lp;
      v8bf hi = *(const v8bf*)(lp + 16);
      pf = pack16(lo, hi);
    }
    // O += P @ V from the shared V^T tile
#pragma unroll
    for (int nt = 0; nt < 4; ++nt) {
      const bf16* vp = &ldsV[cur][(nt * 16 + l16) * 40 + 8 * half];
      v8bf vlo = *(const v8bf*)vp;
      v8bf vhi = *(const v8bf*)(vp + 16);
      v16bf vf = pack16(vlo, vhi);
      o[nt] = __builtin_amdgcn_wmma_f32_16x16x32_bf16(
          false, pf, false, vf, (short)0, o[nt], false, false);
    }
    __syncthreads();
    cur ^= 1;
  }
  // normalize and write ctx (bf16, head-packed)
#pragma unroll
  for (int nt = 0; nt < 4; ++nt)
#pragma unroll
    for (int r = 0; r < 8; ++r) {
      int qg = q0 + r + 8 * half;
      ctx[(size_t)(b * S1 + qg) * 512 + h * 64 + nt * 16 + l16] =
          (bf16)(o[nt][r] / lrow[r]);
    }
}

// ---------------------------------------------------------------------------
// Fused residual add + LayerNorm over D=512. One block (256 thr) per row.
// ---------------------------------------------------------------------------
__global__ __launch_bounds__(256)
void add_ln(const float* __restrict__ y, const float* __restrict__ res,
            const float* __restrict__ g, const float* __restrict__ beta,
            float* __restrict__ outF, bf16* __restrict__ outB, int D) {
  const int row = blockIdx.x;
  __shared__ float red[256];
  float vals[2];
  float local = 0.f;
#pragma unroll
  for (int i = 0; i < 2; ++i) {
    int c = threadIdx.x + i * 256;
    float v = y[(size_t)row * D + c] + res[(size_t)row * D + c];
    vals[i] = v;
    local += v;
  }
  red[threadIdx.x] = local;
  __syncthreads();
  for (int s = 128; s > 0; s >>= 1) {
    if (threadIdx.x < s) red[threadIdx.x] += red[threadIdx.x + s];
    __syncthreads();
  }
  float mean = red[0] / (float)D;
  __syncthreads();
  float lv = 0.f;
#pragma unroll
  for (int i = 0; i < 2; ++i) { float d = vals[i] - mean; lv += d * d; }
  red[threadIdx.x] = lv;
  __syncthreads();
  for (int s = 128; s > 0; s >>= 1) {
    if (threadIdx.x < s) red[threadIdx.x] += red[threadIdx.x + s];
    __syncthreads();
  }
  float inv = rsqrtf(red[0] / (float)D + 1e-5f);
#pragma unroll
  for (int i = 0; i < 2; ++i) {
    int c = threadIdx.x + i * 256;
    float o = (vals[i] - mean) * inv * g[c] + beta[c];
    outF[(size_t)row * D + c] = o;
    if (outB) outB[(size_t)row * D + c] = (bf16)o;
  }
}

// ---------------------------------------------------------------------------
// Exact GELU on f32 input (bias already added by GEMM), bf16 output
// ---------------------------------------------------------------------------
__global__ void gelu_bf16(const float* __restrict__ x, bf16* __restrict__ y, size_t n) {
  size_t i = (size_t)blockIdx.x * blockDim.x + threadIdx.x;
  size_t stride = (size_t)gridDim.x * blockDim.x;
  for (; i < n; i += stride) {
    float v = x[i];
    y[i] = (bf16)(0.5f * v * (1.0f + erff(v * 0.70710678118654752f)));
  }
}

// ---------------------------------------------------------------------------
extern "C" void kernel_launch(void* const* d_in, const int* in_sizes, int n_in,
                              void* d_out, int out_size, void* d_ws, size_t ws_size,
                              hipStream_t stream) {
  const int B = 8, S1 = 768, S2 = 1024, D = 512, DF = 2048;
  const int M1 = B * S1;   // 6144
  const int Msp = B * S2;  // 8192

  const float* cords   = (const float*)d_in[0];
  const float* spatial = (const float*)d_in[1];
  const float* speed   = (const float*)d_in[2];
  const float* wq1 = (const float*)d_in[3];
  const float* wk1 = (const float*)d_in[4];
  const float* wv1 = (const float*)d_in[5];
  const float* wo1 = (const float*)d_in[6];
  const float* bo1 = (const float*)d_in[7];
  const float* wq2 = (const float*)d_in[8];
  const float* wk2 = (const float*)d_in[9];
  const float* wv2 = (const float*)d_in[10];
  const float* wo2 = (const float*)d_in[11];
  const float* bo2 = (const float*)d_in[12];
  const float* ln1g = (const float*)d_in[13];
  const float* ln1b = (const float*)d_in[14];
  const float* ln2g = (const float*)d_in[15];
  const float* ln2b = (const float*)d_in[16];
  const float* ln3g = (const float*)d_in[17];
  const float* ln3b = (const float*)d_in[18];
  const float* fw1 = (const float*)d_in[19];
  const float* fb1 = (const float*)d_in[20];
  const float* fw2 = (const float*)d_in[21];
  const float* fb2 = (const float*)d_in[22];

  char* wsp = (char*)d_ws;
  auto alloc = [&](size_t bytes) -> void* {
    void* p = (void*)wsp;
    wsp += (bytes + 255) & ~(size_t)255;
    return p;
  };

  bf16* cb   = (bf16*)alloc((size_t)M1 * D * 2);
  bf16* spb  = (bf16*)alloc((size_t)Msp * D * 2);
  bf16* sdb  = (bf16*)alloc((size_t)M1 * D * 2);
  bf16* wqb1 = (bf16*)alloc((size_t)D * D * 2);
  bf16* wkb1 = (bf16*)alloc((size_t)D * D * 2);
  bf16* wvb1 = (bf16*)alloc((size_t)D * D * 2);
  bf16* wob1 = (bf16*)alloc((size_t)D * D * 2);
  bf16* wqb2 = (bf16*)alloc((size_t)D * D * 2);
  bf16* wkb2 = (bf16*)alloc((size_t)D * D * 2);
  bf16* wvb2 = (bf16*)alloc((size_t)D * D * 2);
  bf16* wob2 = (bf16*)alloc((size_t)D * D * 2);
  bf16* fw1b = (bf16*)alloc((size_t)DF * D * 2);
  bf16* fw2b = (bf16*)alloc((size_t)D * DF * 2);
  bf16* qb   = (bf16*)alloc((size_t)M1 * D * 2);
  bf16* kb   = (bf16*)alloc((size_t)Msp * D * 2);
  bf16* vb   = (bf16*)alloc((size_t)Msp * D * 2);
  bf16* vt   = (bf16*)alloc((size_t)B * 8 * 64 * S2 * 2);
  bf16* ctx  = (bf16*)alloc((size_t)M1 * D * 2);
  float* gout = (float*)alloc((size_t)M1 * DF * 4);
  float* resA = (float*)alloc((size_t)M1 * D * 4);
  float* resB = (float*)alloc((size_t)M1 * D * 4);
  bf16* xb   = (bf16*)alloc((size_t)M1 * D * 2);
  bf16* hb   = (bf16*)alloc((size_t)M1 * DF * 2);

  auto cvt = [&](const float* src, bf16* dst, size_t n) {
    int blocks = (int)((n + 255) / 256);
    if (blocks > 2048) blocks = 2048;
    cvt_f32_bf16<<<blocks, 256, 0, stream>>>(src, dst, n);
  };

  cvt(cords,   cb,  (size_t)M1 * D);
  cvt(spatial, spb, (size_t)Msp * D);
  cvt(speed,   sdb, (size_t)M1 * D);
  cvt(wq1, wqb1, (size_t)D * D);
  cvt(wk1, wkb1, (size_t)D * D);
  cvt(wv1, wvb1, (size_t)D * D);
  cvt(wo1, wob1, (size_t)D * D);
  cvt(wq2, wqb2, (size_t)D * D);
  cvt(wk2, wkb2, (size_t)D * D);
  cvt(wv2, wvb2, (size_t)D * D);
  cvt(wo2, wob2, (size_t)D * D);
  cvt(fw1, fw1b, (size_t)DF * D);
  cvt(fw2, fw2b, (size_t)D * DF);

  auto gemm = [&](const bf16* A, const bf16* W, const float* bias,
                  float* oF, bf16* oB, int M, int N, int K) {
    dim3 grid(N / 64, M / 256);
    gemm_bf16_wmma<<<grid, 256, 0, stream>>>(A, W, bias, oF, oB, M, N, K);
  };

  // --- attention layer 1 (x = cords, kv = spatial, S2 = 1024, mask mode 1)
  gemm(cb,  wqb1, nullptr, nullptr, qb, M1, D, D);
  gemm(spb, wkb1, nullptr, nullptr, kb, Msp, D, D);
  gemm(spb, wvb1, nullptr, nullptr, vb, Msp, D, D);
  transpose_v<<<2048, 256, 0, stream>>>(vb, vt, S2);
  attn_wmma<<<dim3(S1 / 64, 8, B), 128, 0, stream>>>(qb, kb, vt, ctx, S1, S2, 1);
  gemm(ctx, wob1, bo1, gout, nullptr, M1, D, D);
  add_ln<<<M1, 256, 0, stream>>>(gout, cords, ln1g, ln1b, resA, xb, D);

  // --- attention layer 2 (x = LN1 out, kv = speed, S2 = 768, mask mode 2)
  gemm(xb,  wqb2, nullptr, nullptr, qb, M1, D, D);
  gemm(sdb, wkb2, nullptr, nullptr, kb, M1, D, D);
  gemm(sdb, wvb2, nullptr, nullptr, vb, M1, D, D);
  transpose_v<<<2048, 256, 0, stream>>>(vb, vt, S1);
  attn_wmma<<<dim3(S1 / 64, 8, B), 128, 0, stream>>>(qb, kb, vt, ctx, S1, S1, 2);
  gemm(ctx, wob2, bo2, gout, nullptr, M1, D, D);
  add_ln<<<M1, 256, 0, stream>>>(gout, resA, ln2g, ln2b, resB, xb, D);

  // --- FFN
  gemm(xb, fw1b, fb1, gout, nullptr, M1, DF, D);
  gelu_bf16<<<2048, 256, 0, stream>>>(gout, hb, (size_t)M1 * DF);
  gemm(hb, fw2b, fb2, gout, nullptr, M1, D, DF);
  add_ln<<<M1, 256, 0, stream>>>(gout, resB, ln3g, ln3b, (float*)d_out, nullptr, D);
}
